// DeepSeekMoE_32366873542852
// MI455X (gfx1250) — matrix operand order repaired and checked by reading, not compiled, not run on hardware
//
#include <hip/hip_runtime.h>
#include <hip/hip_bf16.h>

#define T_TOK 4096
#define DM    1024
#define FFD   1024
#define NE    8
#define MM    (1024 * 1024)

typedef __attribute__((ext_vector_type(16))) __bf16 v16bf;
typedef __attribute__((ext_vector_type(8)))  float  v8f;

struct B32bytes { uint4 lo, hi; };

// A fragment (16x32 bf16, wave32): lanes 0-15 hold M=lane, K in {k0..k0+7} (VGPR0-3)
// and {k0+16..k0+23} (VGPR4-7); lanes 16-31 hold the +8 K offsets.
__device__ __forceinline__ v16bf load_a_frag(const __bf16* rowp, int k0, int half) {
  B32bytes s;
  s.lo = *(const uint4*)(rowp + k0 + 8 * half);
  s.hi = *(const uint4*)(rowp + k0 + 16 + 8 * half);
  return __builtin_bit_cast(v16bf, s);
}

// B fragment (32x16 bf16): lanes 0-15 hold N=lane, K in {k0..k0+7} and {k0+16..k0+23};
// lanes 16-31 hold N=lane-16, the +8 K offsets. Weight stored N-major (pre-transposed).
__device__ __forceinline__ v16bf load_b_frag(const __bf16* colp, int k0, int half) {
  B32bytes s;
  const __bf16* p = colp + k0 + 8 * half;
  s.lo = *(const uint4*)(p);
  s.hi = *(const uint4*)(p + 16);
  return __builtin_bit_cast(v16bf, s);
}

__device__ __forceinline__ float gelu_erf(float v) {
  return 0.5f * v * (1.0f + erff(v * 0.70710678118654752f));
}

// ---------------- kernel 1: x fp32 -> bf16 ----------------
__global__ void k_convert_x(const float* __restrict__ x, __bf16* __restrict__ xb) {
  int i = blockIdx.x * 256 + threadIdx.x;
  xb[i] = (__bf16)x[i];
}

// ---------------- kernel 2: transpose+convert all 18 weight matrices ----------------
// z: 0=sw1, 1=sw2, 2..9=w1[e], 10..17=w2[e]. Each is 1024x1024 [K,N] fp32 -> [N,K] bf16.
__global__ void k_transpose_w(const float* __restrict__ sw1, const float* __restrict__ sw2,
                              const float* __restrict__ w1,  const float* __restrict__ w2,
                              __bf16* __restrict__ wb) {
  __shared__ float tile[32][33];
  const int z = blockIdx.z;
  const float* src;
  if (z == 0)      src = sw1;
  else if (z == 1) src = sw2;
  else if (z < 10) src = w1 + (size_t)(z - 2) * MM;
  else             src = w2 + (size_t)(z - 10) * MM;
  __bf16* dst = wb + (size_t)z * MM;

  const int tx = threadIdx.x, ty = threadIdx.y;     // block (32, 8)
  const int c0 = blockIdx.x * 32, r0 = blockIdx.y * 32;
#pragma unroll
  for (int i = 0; i < 4; ++i)
    tile[ty + 8 * i][tx] = src[(size_t)(r0 + ty + 8 * i) * 1024 + c0 + tx];
  __syncthreads();
#pragma unroll
  for (int i = 0; i < 4; ++i)
    dst[(size_t)(c0 + ty + 8 * i) * 1024 + r0 + tx] = (__bf16)tile[tx][ty + 8 * i];
}

// ---------------- kernel 3: gating (softmax over 8 experts + top-2) ----------------
__global__ void k_gate(const float* __restrict__ x, const float* __restrict__ gw,
                       const float* __restrict__ gb,
                       float* __restrict__ topk_p, int* __restrict__ topk_i) {
  const int lane  = threadIdx.x & 31;
  const int wave  = threadIdx.x >> 5;
  const int token = blockIdx.x * 8 + wave;

  float acc[NE];
#pragma unroll
  for (int e = 0; e < NE; ++e) acc[e] = 0.0f;

  const float* xr = x + (size_t)token * DM;
  for (int d = lane; d < DM; d += 32) {
    float xv = xr[d];
    const float* g = gw + (size_t)d * NE;
#pragma unroll
    for (int e = 0; e < NE; ++e) acc[e] += xv * g[e];
  }
#pragma unroll
  for (int e = 0; e < NE; ++e)
    for (int s = 16; s > 0; s >>= 1) acc[e] += __shfl_xor(acc[e], s, 32);

  if (lane == 0) {
    float lg[NE], p[NE];
    float m = -3.4e38f;
#pragma unroll
    for (int e = 0; e < NE; ++e) { lg[e] = acc[e] + gb[e]; m = fmaxf(m, lg[e]); }
    float s = 0.0f;
#pragma unroll
    for (int e = 0; e < NE; ++e) { p[e] = __expf(lg[e] - m); s += p[e]; }
    float inv = 1.0f / s;
    int i0 = 0;
#pragma unroll
    for (int e = 1; e < NE; ++e) if (p[e] > p[i0]) i0 = e;
    int i1 = (i0 == 0) ? 1 : 0;
#pragma unroll
    for (int e = 0; e < NE; ++e) if (e != i0 && p[e] > p[i1]) i1 = e;
    topk_p[token * 2 + 0] = p[i0] * inv;
    topk_p[token * 2 + 1] = p[i1] * inv;
    topk_i[token * 2 + 0] = i0;
    topk_i[token * 2 + 1] = i1;
  }
}

// ---------------- kernel 4: deterministic per-expert token-list compaction ----------------
// Single wave: ballot + prefix popcount, token order preserved (deterministic).
__global__ void k_build_lists(const int* __restrict__ topk_i, int* __restrict__ list,
                              int* __restrict__ meta /*[0..7]=offsets,[8..15]=counts*/,
                              int* __restrict__ assign_pos) {
  const int lane = threadIdx.x;
  int base = 0;
  for (int e = 0; e < NE; ++e) {
    const int start = base;
    if (lane == 0) meta[e] = start;
    for (int t0 = 0; t0 < T_TOK; t0 += 32) {
      const int t  = t0 + lane;
      const int i0 = topk_i[t * 2 + 0];
      const int i1 = topk_i[t * 2 + 1];
      const int km = (i0 == e) ? 0 : ((i1 == e) ? 1 : -1);
      const unsigned mask = __builtin_amdgcn_ballot_w32(km >= 0);
      const int prefix = __popc(mask & ((1u << lane) - 1u));
      if (km >= 0) {
        const int idx = base + prefix;
        list[idx] = t;
        assign_pos[t * 2 + km] = idx;
      }
      base += __popc(mask);
    }
    if (lane == 0) meta[NE + e] = base - start;
  }
}

// ---------------- kernels 5/6: grouped GEMM (WMMA bf16, f32 accum) ----------------
// grid (32, 16, 9), block 128 (4 waves). Wave computes a 32x64 output tile
// (2 M-frags x 4 N-frags): every B fragment feeds 2 WMMAs -> 1.5 B128 loads/WMMA.
// z<8: routed expert z (rows = compact assignment arena); z==8: shared expert (rows = tokens).
template <int STAGE>
__global__ void k_ffn(const __bf16* __restrict__ Ag,   // stage1: xb ; stage2: Hr
                      const __bf16* __restrict__ As,   // stage1: xb ; stage2: Hs
                      const __bf16* __restrict__ wb,
                      const float* __restrict__ bias_r,  // b1[E,FF] / b2[E,D]
                      const float* __restrict__ bias_s,  // sb1 / sb2
                      const int* __restrict__ list, const int* __restrict__ meta,
                      void* __restrict__ Outr, void* __restrict__ Outs) {
  const int z = blockIdx.z;
  int off = 0, cnt = T_TOK;
  if (z < NE) {
    off = meta[z];
    cnt = meta[NE + z];
    if ((int)(blockIdx.x * 128) >= cnt) return;  // uniform exit: EXEC stays all-1s for WMMA
  }
  const int lane = threadIdx.x & 31;
  const int wave = threadIdx.x >> 5;
  const int r    = lane & 15;
  const int half = lane >> 4;
  const int m0   = blockIdx.x * 128 + wave * 32;
  const int n0   = blockIdx.y * 64;

  const int wsel = (STAGE == 1) ? (z == NE ? 0 : 2 + z) : (z == NE ? 1 : 10 + z);
  const __bf16* Wt = wb + (size_t)wsel * MM;

  // two A-fragment row pointers (rows m0+r and m0+16+r), gathered/clamped
  const __bf16* abase[2];
#pragma unroll
  for (int mt = 0; mt < 2; ++mt) {
    const int mrow = m0 + 16 * mt + r;
    if (z == NE) {
      abase[mt] = As + (size_t)mrow * DM;
    } else {
      int lr = mrow;
      if (lr > cnt - 1) lr = cnt - 1;            // clamp tail rows to valid data
      const int arow = (STAGE == 1) ? list[off + lr] : (off + lr);
      abase[mt] = Ag + (size_t)arow * DM;
    }
  }

  v8f acc[2][4];
#pragma unroll
  for (int mt = 0; mt < 2; ++mt)
#pragma unroll
    for (int t = 0; t < 4; ++t)
#pragma unroll
      for (int j = 0; j < 8; ++j) acc[mt][t][j] = 0.0f;

  for (int k0 = 0; k0 < DM; k0 += 32) {
    const v16bf a0 = load_a_frag(abase[0], k0, half);
    const v16bf a1 = load_a_frag(abase[1], k0, half);
#pragma unroll
    for (int t = 0; t < 4; ++t) {
      const __bf16* colp = Wt + (size_t)(n0 + 16 * t + r) * DM;
      const v16bf b = load_b_frag(colp, k0, half);
      acc[0][t] = __builtin_amdgcn_wmma_f32_16x16x32_bf16(
          false, a0, false, b, (short)0, acc[0][t], false, false);
      acc[1][t] = __builtin_amdgcn_wmma_f32_16x16x32_bf16(
          false, a1, false, b, (short)0, acc[1][t], false, false);
    }
  }

  const float* bias = (z == NE) ? bias_s : (bias_r + (size_t)z * 1024);
#pragma unroll
  for (int mt = 0; mt < 2; ++mt) {
#pragma unroll
    for (int t = 0; t < 4; ++t) {
      const int n  = n0 + 16 * t + r;            // C/D layout: N = lane&15
      const float bv = bias[n];
#pragma unroll
      for (int j = 0; j < 8; ++j) {
        const int row = m0 + 16 * mt + 8 * half + j;  // C/D layout: M = j + 8*(lane>>4)
        if (z == NE || row < cnt) {
          float v = acc[mt][t][j] + bv;
          const size_t orow = (z == NE) ? (size_t)row : (size_t)(off + row);
          if (STAGE == 1) {
            v = gelu_erf(v);
            __bf16* outp = (z == NE) ? (__bf16*)Outs : (__bf16*)Outr;
            outp[orow * 1024 + n] = (__bf16)v;
          } else {
            float* outp = (z == NE) ? (float*)Outs : (float*)Outr;
            outp[orow * 1024 + n] = v;
          }
        }
      }
    }
  }
}

// ---------------- kernel 7: combine (shared + weighted routed) ----------------
__global__ void k_combine(const float* __restrict__ Ys, const float* __restrict__ Yr,
                          const float* __restrict__ topk_p, const int* __restrict__ assign_pos,
                          float* __restrict__ out) {
  const int i = blockIdx.x * 256 + threadIdx.x;   // over T*D
  const int t = i >> 10;
  const int d = i & 1023;
  const float p0 = topk_p[t * 2 + 0], p1 = topk_p[t * 2 + 1];
  const int   a0 = assign_pos[t * 2 + 0], a1 = assign_pos[t * 2 + 1];
  out[i] = Ys[i] + p0 * Yr[(size_t)a0 * 1024 + d] + p1 * Yr[(size_t)a1 * 1024 + d];
}

extern "C" void kernel_launch(void* const* d_in, const int* in_sizes, int n_in,
                              void* d_out, int out_size, void* d_ws, size_t ws_size,
                              hipStream_t stream) {
  const float* x      = (const float*)d_in[0];
  const float* gate_w = (const float*)d_in[1];
  const float* gate_b = (const float*)d_in[2];
  const float* sw1    = (const float*)d_in[3];
  const float* sb1    = (const float*)d_in[4];
  const float* sw2    = (const float*)d_in[5];
  const float* sb2    = (const float*)d_in[6];
  const float* w1     = (const float*)d_in[7];
  const float* b1     = (const float*)d_in[8];
  const float* w2     = (const float*)d_in[9];
  const float* b2     = (const float*)d_in[10];
  float* out          = (float*)d_out;

  char* ws = (char*)d_ws;
  size_t o = 0;
  auto take = [&](size_t bytes) { char* p = ws + o; o += (bytes + 255) & ~(size_t)255; return p; };

  __bf16* xb   = (__bf16*)take((size_t)T_TOK * DM * 2);          // 8 MB
  __bf16* wb   = (__bf16*)take((size_t)18 * MM * 2);             // 36 MB
  __bf16* Hr   = (__bf16*)take((size_t)2 * T_TOK * FFD * 2);     // 16 MB (8192 rows)
  __bf16* Hs   = (__bf16*)take((size_t)T_TOK * FFD * 2);         // 8 MB
  float*  Yr   = (float*)take((size_t)2 * T_TOK * DM * 4);       // 32 MB
  float*  Ys   = (float*)take((size_t)T_TOK * DM * 4);           // 16 MB
  float*  tkp  = (float*)take((size_t)T_TOK * 2 * 4);
  int*    tki  = (int*)take((size_t)T_TOK * 2 * 4);
  int*    apos = (int*)take((size_t)T_TOK * 2 * 4);
  int*    list = (int*)take((size_t)2 * T_TOK * 4);
  int*    meta = (int*)take(64);
  (void)ws_size; (void)in_sizes; (void)n_in; (void)out_size;

  // 1) x -> bf16
  k_convert_x<<<(T_TOK * DM) / 256, 256, 0, stream>>>(x, xb);
  // 2) weights -> transposed bf16 arena
  k_transpose_w<<<dim3(32, 32, 18), dim3(32, 8), 0, stream>>>(sw1, sw2, w1, w2, wb);
  // 3) gating
  k_gate<<<T_TOK / 8, 256, 0, stream>>>(x, gate_w, gate_b, tkp, tki);
  // 4) deterministic routing lists
  k_build_lists<<<1, 32, 0, stream>>>(tki, list, meta, apos);
  // 5) FFN1: gelu(X @ W1 + b1) for 8 routed groups + shared
  k_ffn<1><<<dim3(32, 16, 9), 128, 0, stream>>>(xb, xb, wb, b1, sb1, list, meta, Hr, Hs);
  // 6) FFN2: H @ W2 + b2
  k_ffn<2><<<dim3(32, 16, 9), 128, 0, stream>>>(Hr, Hs, wb, b2, sb2, list, meta, Yr, Ys);
  // 7) combine
  k_combine<<<(T_TOK * DM) / 256, 256, 0, stream>>>(Ys, Yr, tkp, apos, out);
}
